// TempTransformer_12627203850252
// MI455X (gfx1250) — matrix operand order repaired and checked by reading, not compiled, not run on hardware
//
#include <hip/hip_runtime.h>
#include <stdint.h>
#include <stddef.h>

// ---------------------------------------------------------------------------
// TempTransformer for MI455X (gfx1250, wave32, WMMA + TDM).
// One workgroup (256 threads = 8 wave32) per row (3040 rows).
// Row state lives in LDS (~126KB dynamic). All GEMMs use
// v_wmma_f32_16x16x32_f16 with weights pre-converted to f16 and pre-swizzled
// into a contiguous-k B-fragment layout so the hot loop is 2x b128 global loads +
// 1 WMMA. The initial strided 32x256 f32 row tile is fetched by the Tensor
// Data Mover (tensor_load_to_lds + s_wait_tensorcnt).
// ---------------------------------------------------------------------------

typedef __attribute__((ext_vector_type(16))) _Float16 v16h;
typedef __attribute__((ext_vector_type(8)))  float    v8f;
typedef __attribute__((ext_vector_type(4)))  uint32_t u32x4;
typedef __attribute__((ext_vector_type(8)))  uint32_t u32x8;

typedef union { v16h v; uint32_t d[8]; } Frag16;

#define NROWS 3040
#define FDIM  32
#define CDIM  256
#define HID   512
#define H3    768

struct KParams {
  const float *x, *pos;
  const float *n1_g[2], *n1_b[2], *qkv_b[2], *proj_b[2];
  const float *n2_g[2], *n2_b[2], *fc1_b[2], *fc2_b[2];
  const float *tn_g, *tn_b, *wm_w, *wm_b, *aw, *ab;
  const uint32_t *w_qkv[2], *w_proj[2], *w_fc1[2], *w_fc2[2];
  float *out;
};

// ---- weight swizzle: f32 (K x N row-major) -> f16 packed in the WMMA
// 32x16 B-fragment layout. dword index = ((kt*Nt + nt)*32 + lane)*8 + reg,
// lane: n = (lane&15), k-half = (lane>>4)*16 ; reg packs k = base+reg*2, +1.
__global__ __launch_bounds__(256) void tt_swz(const float* __restrict__ W,
                                              uint32_t* __restrict__ dst,
                                              int K, int N) {
  int idx = blockIdx.x * 256 + threadIdx.x;
  int total = (K * N) >> 1;
  if (idx >= total) return;
  int reg  = idx & 7;
  int lane = (idx >> 3) & 31;
  int tile = idx >> 8;
  int Nt = N >> 4;
  int kt = tile / Nt, nt = tile - kt * Nt;
  int n  = nt * 16 + (lane & 15);
  int k0 = kt * 32 + ((lane >> 4) << 4) + reg * 2;
  union { _Float16 h[2]; uint32_t u; } cv;
  cv.h[0] = (_Float16)W[(size_t)k0 * N + n];
  cv.h[1] = (_Float16)W[(size_t)(k0 + 1) * N + n];
  dst[idx] = cv.u;
}

// ---- row-wise layernorm over C=256; one wave per row, 8 cols per lane.
__device__ inline void layernorm32(const float* __restrict__ src,
                                   const float* __restrict__ g,
                                   const float* __restrict__ b,
                                   _Float16* dst16, float* dstf, int tid) {
  const int lane = tid & 31, wave = tid >> 5;
  for (int m = wave; m < FDIM; m += 8) {
    const float* rowp = src + m * CDIM;
    float s = 0.f, s2 = 0.f;
#pragma unroll
    for (int i = 0; i < 8; ++i) {
      float v = rowp[lane + 32 * i];
      s += v; s2 += v * v;
    }
#pragma unroll
    for (int off = 16; off > 0; off >>= 1) {
      s  += __shfl_xor(s,  off, 32);
      s2 += __shfl_xor(s2, off, 32);
    }
    float mean = s * (1.0f / 256.0f);
    float var  = s2 * (1.0f / 256.0f) - mean * mean;
    float rs   = rsqrtf(var + 1e-6f);
#pragma unroll
    for (int i = 0; i < 8; ++i) {
      int c = lane + 32 * i;
      float v = (rowp[c] - mean) * rs * g[c] + b[c];
      if (dst16) dst16[m * CDIM + c] = (_Float16)v;
      else       dstf[m * CDIM + c]  = v;
    }
  }
}

// ---- 32xN = (32xK f16 LDS) x (KxN swizzled f16 weights) + bias.
// out16!=null -> write f16 (optionally exact gelu); else accumulate into f32.
__device__ inline void gemm32(const _Float16* __restrict__ A, int K, int N,
                              const uint32_t* __restrict__ W,
                              const float* __restrict__ bias,
                              float* outAdd, _Float16* out16, int gelu,
                              int tid) {
  const int lane = tid & 31, wave = tid >> 5;
  const int ln = lane & 15, lh = lane >> 4;
  const int Nt = N >> 4, T = Nt << 1, Ks = K >> 5;
  for (int t = wave; t < T; t += 8) {
    const int mt = t & 1, nt = t >> 1;
    float bv = bias ? bias[nt * 16 + ln] : 0.0f;
    v8f acc;
#pragma unroll
    for (int r = 0; r < 8; ++r) acc[r] = bv;
    const _Float16* arow = A + (mt * 16 + ln) * K;
    const uint32_t* wbase = W + ((size_t)nt * 32 + lane) * 8;
    for (int kk = 0; kk < Ks; ++kk) {
      Frag16 a, bq;
#pragma unroll
      for (int i = 0; i < 8; ++i) {
        int kp = kk * 32 + (lh << 4) + i * 2;
        a.d[i] = *(const uint32_t*)(arow + kp);
      }
      const uint32_t* wp = wbase + (size_t)kk * Nt * 256;
#pragma unroll
      for (int i = 0; i < 8; ++i) bq.d[i] = wp[i];
      __builtin_prefetch(wp + (size_t)Nt * 256, 0, 0);  // global_prefetch_b8
      acc = __builtin_amdgcn_wmma_f32_16x16x32_f16(false, a.v, false, bq.v,
                                                   (short)0, acc, false, false);
    }
#pragma unroll
    for (int r = 0; r < 8; ++r) {
      int m2 = mt * 16 + lh * 8 + r, n2 = nt * 16 + ln;
      if (out16) {
        float xv = acc[r];
        if (gelu) xv = 0.5f * xv * (1.0f + erff(xv * 0.70710678118f));
        out16[m2 * N + n2] = (_Float16)xv;
      } else {
        outAdd[m2 * N + n2] += acc[r];
      }
    }
  }
}

__global__ __launch_bounds__(256) void tt_main(KParams P) {
  extern __shared__ char smem[];
  float*    h       = (float*)smem;                          // 32*256 f32 32KB
  _Float16* qkvb    = (_Float16*)(smem + 32768);             // 32*768 f16 48KB
  _Float16* lnb     = (_Float16*)(smem + 81920);             // 32*256 f16 16KB
  _Float16* attnb   = (_Float16*)(smem + 98304);             // 32*256 f16 16KB
  _Float16* probs   = (_Float16*)(smem + 114688);            // 2*32*32 f16 4KB
  float*    scoresf = (float*)(smem + 118784);               // 2*32*32 f32 8KB
  float*    red     = (float*)(smem + 126976);               // 512 f32 2KB

  const int tid  = threadIdx.x;
  const int row  = blockIdx.x;
  const int bidx = row / 95, pidx = row - bidx * 95;
  const int lane = tid & 31, wave = tid >> 5;
  const int ln = lane & 15, lh = lane >> 4;

  // ---- TDM: DMA the strided row tile x[b, 0:32, p, 0:256] into LDS `h`.
  // D# group0: count=1, lds_addr=0 (h is at LDS offset 0), global tile addr,
  // type=2. group1: data_size=4B, tensor_dim0=256, tensor_dim1=32,
  // tile_dim0=256, tile_dim1=32, tensor_dim0_stride=95*256=24320.
  if (tid < 32) {  // wave 0 only: TDM ignores EXEC, one issue per wave
    const float* gbase = P.x + (((size_t)(bidx * FDIM)) * 95 + pidx) * CDIM;
    uint64_t ga = (uint64_t)(uintptr_t)gbase;
    u32x4 g0;
    g0[0] = 1u;                                    // count=1, user descriptor
    g0[1] = 0u;                                    // lds_addr (bytes)
    g0[2] = (uint32_t)ga;                          // global_addr[31:0]
    g0[3] = (uint32_t)((ga >> 32) & 0x01FFFFFFu) | (2u << 30);  // addr|type=2
    u32x8 g1;
    g1[0] = (2u << 16);        // workgroup_mask=0, data_size=2 (4 bytes)
    g1[1] = (256u << 16);      // tensor_dim0[15:0] at bits 63:48
    g1[2] = (32u << 16);       // tensor_dim0 hi=0 | tensor_dim1[15:0]=32
    g1[3] = (256u << 16);      // tensor_dim1 hi=0 | tile_dim0=256
    g1[4] = 32u;               // tile_dim1=32, tile_dim2=0
    g1[5] = 24320u;            // tensor_dim0_stride[31:0]
    g1[6] = 0u;
    g1[7] = 0u;
    asm volatile("tensor_load_to_lds %0, %1" :: "s"(g0), "s"(g1) : "memory");
    __builtin_amdgcn_s_wait_tensorcnt(0);
  }
  __syncthreads();

  // ---- add positional embedding in place
  for (int f = 0; f < FDIM; ++f)
    h[f * CDIM + tid] += P.pos[f * CDIM + tid];
  __syncthreads();

  for (int blk = 0; blk < 2; ++blk) {
    // ---- LN1 -> lnb(f16), QKV GEMM -> qkvb(f16, 32x768 = [q|k|v])
    layernorm32(h, P.n1_g[blk], P.n1_b[blk], lnb, nullptr, tid);
    __syncthreads();
    gemm32(lnb, CDIM, H3, P.w_qkv[blk], P.qkv_b[blk], nullptr, qkvb, 0, tid);
    __syncthreads();

    // ---- scores = q @ k^T * scale : per head M=N=32, K=128 (8 tiles, 1/wave)
    {
      const int hg = wave >> 2, t = wave & 3, mt = t & 1, nt = t >> 1;
      v8f acc;
#pragma unroll
      for (int r = 0; r < 8; ++r) acc[r] = 0.0f;
      const _Float16* arow = qkvb + (mt * 16 + ln) * H3 + hg * 128;
      const _Float16* brow = qkvb + (nt * 16 + ln) * H3 + 256 + hg * 128 + (lh << 4);
      for (int kk = 0; kk < 4; ++kk) {
        Frag16 a, bq;
#pragma unroll
        for (int i = 0; i < 8; ++i) {
          int kp = kk * 32 + (lh << 4) + i * 2;
          a.d[i]  = *(const uint32_t*)(arow + kp);
          bq.d[i] = *(const uint32_t*)(brow + kk * 32 + i * 2);
        }
        acc = __builtin_amdgcn_wmma_f32_16x16x32_f16(false, a.v, false, bq.v,
                                                     (short)0, acc, false, false);
      }
#pragma unroll
      for (int r = 0; r < 8; ++r)
        scoresf[(hg * 32 + mt * 16 + lh * 8 + r) * 32 + nt * 16 + ln] =
            acc[r] * 0.08838834764831845f;  // 128^-0.5
    }
    __syncthreads();

    // ---- softmax over 64 rows (2 heads x 32)
    if (tid < 64) {
      float* rowp = scoresf + tid * 32;
      float mx = rowp[0];
      for (int n = 1; n < 32; ++n) mx = fmaxf(mx, rowp[n]);
      float sum = 0.f;
      for (int n = 0; n < 32; ++n) { float e = __expf(rowp[n] - mx); rowp[n] = e; sum += e; }
      float inv = 1.0f / sum;
      for (int n = 0; n < 32; ++n) probs[tid * 32 + n] = (_Float16)(rowp[n] * inv);
    }
    __syncthreads();

    // ---- out = probs @ v : per head M=32, K=32, N=128 (32 tiles, 4/wave)
    for (int t = wave; t < 32; t += 8) {
      const int hg = t >> 4, rt = t & 15, mt = rt & 1, nt = rt >> 1;
      v8f acc;
#pragma unroll
      for (int r = 0; r < 8; ++r) acc[r] = 0.0f;
      Frag16 a, bq;
      const _Float16* arow = probs + (hg * 32 + mt * 16 + ln) * 32;
#pragma unroll
      for (int i = 0; i < 8; ++i) {
        int kp = (lh << 4) + i * 2;
        a.d[i] = *(const uint32_t*)(arow + kp);
      }
      const int coln = 512 + hg * 128 + nt * 16 + ln;
#pragma unroll
      for (int i = 0; i < 8; ++i) {
        int k0 = (lh << 4) + i * 2;
        union { _Float16 h2[2]; uint32_t u; } pk;
        pk.h2[0] = qkvb[k0 * H3 + coln];
        pk.h2[1] = qkvb[(k0 + 1) * H3 + coln];
        bq.d[i] = pk.u;
      }
      acc = __builtin_amdgcn_wmma_f32_16x16x32_f16(false, a.v, false, bq.v,
                                                   (short)0, acc, false, false);
#pragma unroll
      for (int r = 0; r < 8; ++r)
        attnb[(mt * 16 + lh * 8 + r) * CDIM + hg * 128 + nt * 16 + ln] =
            (_Float16)acc[r];
    }
    __syncthreads();

    // ---- proj (+residual into h)
    gemm32(attnb, CDIM, CDIM, P.w_proj[blk], P.proj_b[blk], h, nullptr, 0, tid);
    __syncthreads();

    // ---- LN2 -> lnb, FC1+gelu -> qkvb (32x512), FC2 (+residual into h)
    layernorm32(h, P.n2_g[blk], P.n2_b[blk], lnb, nullptr, tid);
    __syncthreads();
    gemm32(lnb, CDIM, HID, P.w_fc1[blk], P.fc1_b[blk], nullptr, qkvb, 1, tid);
    __syncthreads();
    gemm32(qkvb, HID, CDIM, P.w_fc2[blk], P.fc2_b[blk], h, nullptr, 0, tid);
    __syncthreads();
  }

  // ---- final layernorm (in place, f32)
  layernorm32(h, P.tn_g, P.tn_b, nullptr, h, tid);
  __syncthreads();

  // ---- x_b = einsum(h, wm_w) + wm_b ; s2 = dot(x_b, aw2) + ab
  {
    float s = 0.f;
    for (int f = 0; f < FDIM; ++f) s += h[f * CDIM + tid] * P.wm_w[f];
    float xb = s + P.wm_b[0];
    red[tid] = xb * P.aw[CDIM + tid];
  }
  __syncthreads();
  for (int off = 128; off > 0; off >>= 1) {
    if (tid < off) red[tid] += red[tid + off];
    __syncthreads();
  }
  float sglob = red[0] + P.ab[0];
  if (tid < FDIM) {
    float s = 0.f;
    for (int c = 0; c < CDIM; ++c) s += h[tid * CDIM + c] * P.aw[c];
    red[256 + tid] = s + sglob;  // alpha[f]
  }
  __syncthreads();
  float den = 0.f;
  for (int f = 0; f < FDIM; ++f) den += red[256 + f];
  {
    float s = 0.f;
    for (int f = 0; f < FDIM; ++f) s += h[f * CDIM + tid] * red[256 + f];
    P.out[(size_t)row * CDIM + tid] = s / den;
  }
}

// ---------------------------------------------------------------------------
extern "C" void kernel_launch(void* const* d_in, const int* in_sizes, int n_in,
                              void* d_out, int out_size, void* d_ws,
                              size_t ws_size, hipStream_t stream) {
  (void)n_in; (void)out_size; (void)ws_size;
  const float* in[32];
  for (int i = 0; i < 32; ++i) in[i] = (const float*)d_in[i];

  KParams P{};
  const float *qkvW[2], *projW[2], *fc1W[2], *fc2W[2];

  // Two plausible flattening conventions; pick by where the big x tensor is.
  bool xFirst = (in_sizes[0] == 32 * 32 * 95 * 256);
  if (xFirst) {  // dict-insertion order
    P.x = in[0]; P.pos = in[1];
    for (int b = 0; b < 2; ++b) {
      const float* const* q = in + 2 + b * 12;
      P.n1_g[b] = q[0];  P.n1_b[b] = q[1];
      qkvW[b]   = q[2];  P.qkv_b[b] = q[3];
      projW[b]  = q[4];  P.proj_b[b] = q[5];
      P.n2_g[b] = q[6];  P.n2_b[b] = q[7];
      fc1W[b]   = q[8];  P.fc1_b[b] = q[9];
      fc2W[b]   = q[10]; P.fc2_b[b] = q[11];
    }
    P.tn_g = in[26]; P.tn_b = in[27]; P.wm_w = in[28];
    P.wm_b = in[29]; P.aw = in[30];  P.ab = in[31];
  } else {  // JAX pytree sorted-key order
    P.ab = in[0]; P.aw = in[1];
    for (int b = 0; b < 2; ++b) {
      const float* const* q = in + 2 + b * 12;
      P.fc1_b[b] = q[0];  fc1W[b] = q[1];
      P.fc2_b[b] = q[2];  fc2W[b] = q[3];
      P.n1_b[b] = q[4];   P.n1_g[b] = q[5];
      P.n2_b[b] = q[6];   P.n2_g[b] = q[7];
      P.proj_b[b] = q[8]; projW[b] = q[9];
      P.qkv_b[b] = q[10]; qkvW[b] = q[11];
    }
    P.pos = in[26]; P.tn_b = in[27]; P.tn_g = in[28];
    P.wm_b = in[29]; P.wm_w = in[30]; P.x = in[31];
  }

  // ---- workspace layout (dwords): per block 256K dwords (1MB)
  uint32_t* ws = (uint32_t*)d_ws;
  for (int b = 0; b < 2; ++b) {
    uint32_t* base = ws + (size_t)b * 262144;
    P.w_qkv[b]  = base;
    P.w_proj[b] = base + 98304;
    P.w_fc1[b]  = base + 131072;
    P.w_fc2[b]  = base + 196608;
  }

  // ---- pre-swizzle weights into WMMA B-fragment layout (f16)
  struct Job { const float* W; uint32_t* dst; int K, N; };
  Job jobs[8];
  for (int b = 0; b < 2; ++b) {
    uint32_t* base = ws + (size_t)b * 262144;
    jobs[b * 4 + 0] = {qkvW[b],  base,          256, 768};
    jobs[b * 4 + 1] = {projW[b], base + 98304,  256, 256};
    jobs[b * 4 + 2] = {fc1W[b],  base + 131072, 256, 512};
    jobs[b * 4 + 3] = {fc2W[b],  base + 196608, 512, 256};
  }
  for (int j = 0; j < 8; ++j) {
    int total = (jobs[j].K * jobs[j].N) >> 1;
    tt_swz<<<(total + 255) / 256, 256, 0, stream>>>(jobs[j].W, jobs[j].dst,
                                                    jobs[j].K, jobs[j].N);
  }

  P.out = (float*)d_out;
  tt_main<<<NROWS, 256, 129024, stream>>>(P);
}
